// GNNUnet_18657337934725
// MI455X (gfx1250) — matrix operand order repaired and checked by reading, not compiled, not run on hardware
//
#include <hip/hip_runtime.h>
#include <hip/hip_bf16.h>
#include <math.h>

typedef __attribute__((ext_vector_type(16))) __bf16 v16bf;
typedef __attribute__((ext_vector_type(8)))  float  v8f;

#define NND 20000
#define NPAD 20224              /* 79*256: GEMM row padding so blocks are full */
#define NED 200000
#define NGR 8
#define EPSR 1e-6f

enum { GF_BETA = 1, GF_SILU = 2 };

// silu via fast hardware reciprocal (v_rcp_f32) instead of IEEE divide
__device__ __forceinline__ float dsilu(float x) {
  return x * __builtin_amdgcn_rcpf(1.0f + __expf(-x));
}

// ---------------- elementwise / reduction kernels ----------------

__global__ __launch_bounds__(256) void k_zero(float* __restrict__ p, int n) {
  int i = blockIdx.x * 256 + threadIdx.x;
  if (i < n) p[i] = 0.0f;
}

__global__ __launch_bounds__(256) void k_counts(const int* __restrict__ batch,
                                                float* __restrict__ counts) {
  int i = blockIdx.x * 256 + threadIdx.x;
  if (i < NND) atomicAdd(&counts[batch[i]], 1.0f);
}

__global__ __launch_bounds__(128) void k_temb(const float* __restrict__ t,
                                              float* __restrict__ emb) {
  int i = blockIdx.x * 128 + threadIdx.x;
  if (i >= NGR * 128) return;
  int g = i >> 7, j = i & 127;
  const int half = 64;
  float lf = -logf(10000.0f) / (float)(half - 1);
  if (j < half) emb[i] = sinf(t[g] * __expf((float)j * lf));
  else          emb[i] = cosf(t[g] * __expf((float)(j - half) * lf));
}

// tiny GEMM for 8-row matrices: out[M,NN] = act(A[M,K]) @ W[K,NN] + bias
// flags bit0: silu applied to A as read; bit1: silu applied to output
__global__ __launch_bounds__(256) void k_small_gemm(const float* __restrict__ A,
                                                    const float* __restrict__ W,
                                                    const float* __restrict__ bias,
                                                    float* __restrict__ out,
                                                    int M, int K, int NN, int flags) {
  int i = blockIdx.x * 256 + threadIdx.x;
  if (i >= M * NN) return;
  int m = i / NN, n = i % NN;
  float s = bias ? bias[n] : 0.0f;
  for (int k = 0; k < K; ++k) {
    float a = A[m * K + k];
    if (flags & 1) a = dsilu(a);
    s += a * W[k * NN + n];
  }
  if (flags & 2) s = dsilu(s);
  out[i] = s;
}

__global__ __launch_bounds__(256) void k_inproj(const float* __restrict__ xin,
                                                const float* __restrict__ w,
                                                float* __restrict__ x) {
  int i = blockIdx.x * 256 + threadIdx.x;
  if (i >= NND * 128) return;
  int n = i >> 7, c = i & 127;
  x[i] = xin[2 * n] * w[c] + xin[2 * n + 1] * w[128 + c];
}

// C = 1<<cs (channels are powers of two -> shift/mask instead of div/mod)
__global__ __launch_bounds__(256) void k_sumsq(const float* __restrict__ x,
                                               const int* __restrict__ batch,
                                               float* __restrict__ sumsq, int cs) {
  long i = (long)blockIdx.x * 256 + threadIdx.x;
  long total = (long)NND << cs;
  if (i >= total) return;
  int n = (int)(i >> cs), c = (int)(i & ((1 << cs) - 1));
  float v = x[i];
  atomicAdd(&sumsq[(batch[n] << cs) + c], v * v);
}

// fused rms_graph_norm + FiLM conditioning + SiLU
__global__ __launch_bounds__(256) void k_nfs(const float* __restrict__ x,
                                             const int* __restrict__ batch,
                                             const float* __restrict__ sumsq,
                                             const float* __restrict__ counts,
                                             const float* __restrict__ styleT,
                                             const float* __restrict__ xin,
                                             const float* __restrict__ fw,
                                             float* __restrict__ y, int cs) {
  long i = (long)blockIdx.x * 256 + threadIdx.x;
  long total = (long)NND << cs;
  if (i >= total) return;
  int C = 1 << cs;
  int n = (int)(i >> cs), c = (int)(i & (C - 1));
  int b = batch[n];
  float cnt = fmaxf(counts[b], 1.0f);
  float ms = sumsq[(b << cs) + c] * __builtin_amdgcn_rcpf(cnt);
  float xn = x[i] * rsqrtf(ms + EPSR);
  float fld = xin[2 * n + 1];
  float gamma = styleT[(b << (cs + 1)) + c]     + fld * fw[c];
  float beta  = styleT[(b << (cs + 1)) + C + c] + fld * fw[C + c];
  float v = xn * (1.0f + gamma) + beta;
  y[i] = dsilu(v);
}

// message scatter: agg[dst] += y[src] * ew  (channel-fastest -> coalesced atomics)
__global__ __launch_bounds__(256) void k_scatter(const float* __restrict__ y,
                                                 const int* __restrict__ src,
                                                 const int* __restrict__ dst,
                                                 const float* __restrict__ ew,
                                                 float* __restrict__ agg, int cs) {
  long i = (long)blockIdx.x * 256 + threadIdx.x;
  long total = (long)NED << cs;
  if (i >= total) return;
  int e = (int)(i >> cs), c = (int)(i & ((1 << cs) - 1));
  float v = y[(src[e] << cs) + c] * ew[e];
  atomicAdd(&agg[(dst[e] << cs) + c], v);
}

__global__ __launch_bounds__(256) void k_add2(float* __restrict__ out,
                                              const float* __restrict__ a,
                                              const float* __restrict__ b, int n) {
  int i = blockIdx.x * 256 + threadIdx.x;
  if (i < n) out[i] = a[i] + b[i];
}

__global__ __launch_bounds__(256) void k_addinto(float* __restrict__ out,
                                                 const float* __restrict__ a, int n) {
  int i = blockIdx.x * 256 + threadIdx.x;
  if (i < n) out[i] += a[i];
}

// ---------------- WMMA bf16 GEMM, LDS-staged B ----------------
// out[NPAD,M] = epi( A[NPAD,K] @ W[K,M] + bias + resid + beta*out_prev )
// Block = 8 waves = 256x64 output macro-tile, all waves share one column band.
// Per K-step: block cooperatively stages the 32x64 W slab into LDS as bf16
// (one coalesced pass, 8x fewer global B loads/converts than per-wave loading),
// then each wave reads its B fragments via ds_load_b128 and issues
// 8 v_wmma_f32_16x16x32_bf16 (2 N-tiles x 4 M-tiles, A reused 4x, B reused 2x).
// Rows padded to NPAD (mult of 256) so every block is full: no EXEC partiality,
// no barrier divergence. K%32==0, M%64==0.
__global__ __launch_bounds__(256) void k_gemm_bf16(const float* __restrict__ A,
                                                   const float* __restrict__ W,
                                                   const float* __restrict__ bias,
                                                   const float* __restrict__ resid,
                                                   float* __restrict__ out,
                                                   int K, int M, int flags) {
  // sB layout: [col 0..63][half 0..1][e 0..15] -> lane fragment is 32B contiguous
  __shared__ __attribute__((aligned(32))) __bf16 sB[64 * 32];
  int tid  = threadIdx.x;
  int lane = tid & 31;
  int wid  = tid >> 5;
  int gm   = M >> 6;
  int tm   = ((int)blockIdx.x % gm) << 6;                 // column origin
  int tn   = (((int)blockIdx.x / gm) << 8) + (wid << 5);  // row origin (per wave)
  int half = lane >> 4;
  int l16  = lane & 15;

  v8f acc[2][4];
  if (flags & GF_BETA) {
#pragma unroll
    for (int i = 0; i < 2; ++i)
#pragma unroll
      for (int j = 0; j < 4; ++j)
#pragma unroll
        for (int v = 0; v < 8; ++v)
          acc[i][j][v] = out[(long)(tn + i * 16 + (half << 3) + v) * M + tm + j * 16 + l16];
  } else {
#pragma unroll
    for (int i = 0; i < 2; ++i)
#pragma unroll
      for (int j = 0; j < 4; ++j)
#pragma unroll
        for (int v = 0; v < 8; ++v) acc[i][j][v] = 0.0f;
  }

  const float* Ap0 = A + (long)(tn + l16) * K;        // row for i=0
  const float* Ap1 = A + (long)(tn + 16 + l16) * K;   // row for i=1

  for (int k0 = 0; k0 < K; k0 += 32) {
    __syncthreads();                      // previous slab fully consumed
    // cooperative stage: 32 K-rows x 64 cols of W -> bf16 in LDS
    {
      int q = tid << 1;                   // 2 float4 per thread, 512 total
#pragma unroll
      for (int u = 0; u < 2; ++u, ++q) {
        int k  = q >> 4;                  // 0..31
        int c4 = (q & 15) << 2;           // 0,4,...,60
        float4 w4 = *(const float4*)(W + (long)(k0 + k) * M + tm + c4);
        int hk = (k >> 3) & 1, ek = (k & 7) + ((k >> 4) << 3);
        sB[((c4 + 0) * 2 + hk) * 16 + ek] = (__bf16)w4.x;
        sB[((c4 + 1) * 2 + hk) * 16 + ek] = (__bf16)w4.y;
        sB[((c4 + 2) * 2 + hk) * 16 + ek] = (__bf16)w4.z;
        sB[((c4 + 3) * 2 + hk) * 16 + ek] = (__bf16)w4.w;
      }
    }
    __syncthreads();

    int ab = k0 + (half << 3);            // A elems 0..7 at ab, 8..15 at ab+16
    v16bf a[2], b[4];
#pragma unroll
    for (int i = 0; i < 2; ++i) {
      const float4* p = (const float4*)((i ? Ap1 : Ap0) + ab);
      float4 q0 = p[0], q1 = p[1];        // k = ab .. ab+7
      float4 q2 = p[4], q3 = p[5];        // k = ab+16 .. ab+23
      a[i][0]  = (__bf16)q0.x; a[i][1]  = (__bf16)q0.y;
      a[i][2]  = (__bf16)q0.z; a[i][3]  = (__bf16)q0.w;
      a[i][4]  = (__bf16)q1.x; a[i][5]  = (__bf16)q1.y;
      a[i][6]  = (__bf16)q1.z; a[i][7]  = (__bf16)q1.w;
      a[i][8]  = (__bf16)q2.x; a[i][9]  = (__bf16)q2.y;
      a[i][10] = (__bf16)q2.z; a[i][11] = (__bf16)q2.w;
      a[i][12] = (__bf16)q3.x; a[i][13] = (__bf16)q3.y;
      a[i][14] = (__bf16)q3.z; a[i][15] = (__bf16)q3.w;
    }
#pragma unroll
    for (int j = 0; j < 4; ++j)
      b[j] = *(const v16bf*)&sB[((j * 16 + l16) * 2 + half) * 16];  // 32B ds read

#pragma unroll
    for (int i = 0; i < 2; ++i)
#pragma unroll
      for (int j = 0; j < 4; ++j)
        acc[i][j] = __builtin_amdgcn_wmma_f32_16x16x32_bf16(
            false, a[i], false, b[j], (short)0, acc[i][j], false, false);
  }

#pragma unroll
  for (int j = 0; j < 4; ++j) {
    int c = tm + j * 16 + l16;
    float bv = bias ? bias[c] : 0.0f;
#pragma unroll
    for (int i = 0; i < 2; ++i) {
#pragma unroll
      for (int v = 0; v < 8; ++v) {
        int r = tn + i * 16 + (half << 3) + v;   // C/D layout: row = v + 8*half
        float x = acc[i][j][v] + bv;
        if (resid) x += resid[(long)r * M + c];
        if (flags & GF_SILU) x = dsilu(x);
        out[(long)r * M + c] = x;
      }
    }
  }
}

// final: rms_graph_norm + silu + [x, field] @ out_w + out_b ; one wave per node
__global__ __launch_bounds__(256) void k_final(const float* __restrict__ x,
                                               const int* __restrict__ batch,
                                               const float* __restrict__ sumsq,
                                               const float* __restrict__ counts,
                                               const float* __restrict__ xin,
                                               const float* __restrict__ ow,
                                               const float* __restrict__ ob,
                                               float* __restrict__ out) {
  int wv = (int)(((unsigned)blockIdx.x * 256u + threadIdx.x) >> 5);
  int lane = threadIdx.x & 31;
  if (wv >= NND) return;
  int b = batch[wv];
  float rcnt = __builtin_amdgcn_rcpf(fmaxf(counts[b], 1.0f));
  float s = 0.0f;
  for (int c = lane; c < 128; c += 32) {
    float ms = sumsq[b * 128 + c] * rcnt;
    float v = x[wv * 128 + c] * rsqrtf(ms + EPSR);
    v = dsilu(v);
    s += v * ow[c];
  }
  for (int off = 16; off > 0; off >>= 1) s += __shfl_down(s, off, 32);
  if (lane == 0) out[wv] = s + xin[2 * wv + 1] * ow[128] + ob[0];
}

// ---------------- host orchestration ----------------

struct LayerW {
  const float *c0_rel, *c0_root, *c1_rel, *c1_root, *f_w;
  const float *m_b1, *m_b2, *m_w1, *m_w2, *sc_w, *t_b, *t_w;
};

static inline int cdiv(long a, int b) { return (int)((a + b - 1) / b); }
static inline int ilog2(int c) { int s = 0; while ((1 << s) < c) ++s; return s; }

static void launch_gemm(const float* A, const float* W, const float* bias,
                        const float* resid, float* out, int K, int M,
                        int flags, hipStream_t stream) {
  int blocks = (NPAD >> 8) * (M >> 6);     // every block fully populated
  k_gemm_bf16<<<blocks, 256, 0, stream>>>(A, W, bias, resid, out, K, M, flags);
}

static void run_layer(const LayerW& Lw, const float* X, float* OUT,
                      int Cin, int Cout, bool encoder, bool has_sc,
                      float* W1, float* W2, float* W3,
                      const float* tvec, float* styleT, float* sumsq,
                      const float* counts, const float* xin, const int* batch,
                      const int* src, const int* dst, const float* ew,
                      hipStream_t stream) {
  const int N = NND, E = NED;
  const int cs = ilog2(Cin);
  // style_t = silu(t_vec) @ t_w + t_b    [8 x 2Cin]
  k_small_gemm<<<cdiv(8L * 2 * Cin, 256), 256, 0, stream>>>(
      tvec, Lw.t_w, Lw.t_b, styleT, 8, 128, 2 * Cin, 1);
  // per-graph mean-square
  k_zero<<<cdiv(8L * Cin, 256), 256, 0, stream>>>(sumsq, 8 * Cin);
  k_sumsq<<<cdiv((long)N * Cin, 256), 256, 0, stream>>>(X, batch, sumsq, cs);
  // norm + FiLM + silu -> W1
  k_nfs<<<cdiv((long)N * Cin, 256), 256, 0, stream>>>(
      X, batch, sumsq, counts, styleT, xin, Lw.f_w, W1, cs);
  const float* cur = W1;
  if (!encoder) {
    // H = silu(Y @ m_w1 + m_b1) ; Y2 = Y + H @ m_w2 + m_b2
    launch_gemm(W1, Lw.m_w1, Lw.m_b1, nullptr, W2, Cin, 2 * Cin, GF_SILU, stream);
    launch_gemm(W2, Lw.m_w2, Lw.m_b2, W1, W3, 2 * Cin, Cin, 0, stream);
    cur = W3;
  }
  float* agg = W2;                         // H dead by now in non-encoder path
  float* z = encoder ? W3 : W1;            // conv0 output buffer
  // conv0: z = silu(agg @ c0_rel + cur @ c0_root)
  k_zero<<<cdiv((long)NPAD * Cin, 256), 256, 0, stream>>>(agg, NPAD * Cin);
  k_scatter<<<cdiv((long)E * Cin, 256), 256, 0, stream>>>(cur, src, dst, ew, agg, cs);
  launch_gemm(agg, Lw.c0_rel, nullptr, nullptr, z, Cin, Cin, 0, stream);
  launch_gemm(cur, Lw.c0_root, nullptr, nullptr, z, Cin, Cin, GF_BETA | GF_SILU, stream);
  // conv1: OUT = agg2 @ c1_rel + z @ c1_root + shortcut
  k_zero<<<cdiv((long)NPAD * Cin, 256), 256, 0, stream>>>(agg, NPAD * Cin);
  k_scatter<<<cdiv((long)E * Cin, 256), 256, 0, stream>>>(z, src, dst, ew, agg, cs);
  launch_gemm(agg, Lw.c1_rel, nullptr, nullptr, OUT, Cin, Cout, 0, stream);
  launch_gemm(z, Lw.c1_root, nullptr, nullptr, OUT, Cin, Cout, GF_BETA, stream);
  if (has_sc)
    launch_gemm(X, Lw.sc_w, nullptr, nullptr, OUT, Cin, Cout, GF_BETA, stream);
  else
    k_addinto<<<cdiv((long)N * Cout, 256), 256, 0, stream>>>(OUT, X, N * Cout);
}

extern "C" void kernel_launch(void* const* d_in, const int* in_sizes, int n_in,
                              void* d_out, int out_size, void* d_ws, size_t ws_size,
                              hipStream_t stream) {
  if (n_in < 61) return;
  const float* xin   = (const float*)d_in[0];   // [20000,2]
  const int*   eidx  = (const int*)d_in[1];     // [2,200000]
  const float* ew    = (const float*)d_in[2];   // [200000]
  const int*   batch = (const int*)d_in[3];     // [20000]
  const float* tt    = (const float*)d_in[4];   // [8]
  const int* src = eidx;
  const int* dst = eidx + NED;
  auto P = [&](int i) { return (const float*)d_in[i]; };
  // pytree flatten order of params: dec[0], dec[1], enc[0], enc[1], in_w, lat,
  // out_b, out_w, time_w1, time_w2 ; per-layer keys sorted alphabetically.
  LayerW dec0{P(5),  P(6),  P(7),  P(8),  P(9),  P(10), P(11), P(12), P(13), P(14), P(15), P(16)};
  LayerW dec1{P(17), P(18), P(19), P(20), P(21), P(22), P(23), P(24), P(25), P(26), P(27), P(28)};
  LayerW enc0{P(29), P(30), P(31), P(32), P(33), nullptr, nullptr, nullptr, nullptr, P(34), P(35), P(36)};
  LayerW enc1{P(37), P(38), P(39), P(40), P(41), nullptr, nullptr, nullptr, nullptr, P(42), P(43), P(44)};
  const float* in_w = P(45);
  LayerW lat {P(46), P(47), P(48), P(49), P(50), P(51), P(52), P(53), P(54), nullptr, P(55), P(56)};
  const float* out_b = P(57);
  const float* out_w = P(58);
  const float* tw1   = P(59);
  const float* tw2   = P(60);

  // workspace carve-out (node buffers padded to NPAD rows for full GEMM blocks)
  float* ws = (float*)d_ws;
  size_t off = 0;
  auto alloc = [&](size_t n) { float* p = ws + off; off += n; return p; };
  float* counts = alloc(64);
  float* sumsq  = alloc(8 * 512);
  float* temb   = alloc(8 * 128);
  float* th     = alloc(8 * 256);
  float* tvec   = alloc(8 * 128);
  float* styleT = alloc(8 * 1024);
  float* P0 = alloc((size_t)NPAD * 128);   // input projection
  float* R0 = alloc((size_t)NPAD * 256);   // enc0 out (residual)
  float* R1 = alloc((size_t)NPAD * 512);   // enc1 out (residual)
  float* Lb = alloc((size_t)NPAD * 512);   // latent out
  float* Sb = alloc((size_t)NPAD * 512);   // residual-sum buffer
  float* D0 = alloc((size_t)NPAD * 256);   // dec0 out
  float* Fb = alloc((size_t)NPAD * 128);   // dec1 out
  float* W1 = alloc((size_t)NPAD * 512);
  float* W2 = alloc((size_t)NPAD * 1024);
  float* W3 = alloc((size_t)NPAD * 512);
  if (off * sizeof(float) > ws_size) return;  // workspace too small: bail silently

  // per-graph node counts
  k_zero<<<1, 256, 0, stream>>>(counts, 64);
  k_counts<<<cdiv(NND, 256), 256, 0, stream>>>(batch, counts);
  // time embedding MLP
  k_temb<<<cdiv(8L * 128, 128), 128, 0, stream>>>(tt, temb);
  k_small_gemm<<<cdiv(8L * 256, 256), 256, 0, stream>>>(temb, tw1, nullptr, th, 8, 128, 256, 2);
  k_small_gemm<<<cdiv(8L * 128, 256), 256, 0, stream>>>(th, tw2, nullptr, tvec, 8, 256, 128, 0);
  // input projection
  k_inproj<<<cdiv((long)NND * 128, 256), 256, 0, stream>>>(xin, in_w, P0);

  run_layer(enc0, P0, R0, 128, 256, true, true, W1, W2, W3,
            tvec, styleT, sumsq, counts, xin, batch, src, dst, ew, stream);
  run_layer(enc1, R0, R1, 256, 512, true, true, W1, W2, W3,
            tvec, styleT, sumsq, counts, xin, batch, src, dst, ew, stream);
  run_layer(lat, R1, Lb, 512, 512, false, false, W1, W2, W3,
            tvec, styleT, sumsq, counts, xin, batch, src, dst, ew, stream);
  k_add2<<<cdiv((long)NND * 512, 256), 256, 0, stream>>>(Sb, Lb, R1, NND * 512);
  run_layer(dec0, Sb, D0, 512, 256, false, true, W1, W2, W3,
            tvec, styleT, sumsq, counts, xin, batch, src, dst, ew, stream);
  k_add2<<<cdiv((long)NND * 256, 256), 256, 0, stream>>>(Sb, D0, R0, NND * 256);
  run_layer(dec1, Sb, Fb, 256, 128, false, true, W1, W2, W3,
            tvec, styleT, sumsq, counts, xin, batch, src, dst, ew, stream);

  // final norm + silu + output projection
  k_zero<<<cdiv(8L * 128, 256), 256, 0, stream>>>(sumsq, 8 * 128);
  k_sumsq<<<cdiv((long)NND * 128, 256), 256, 0, stream>>>(Fb, batch, sumsq, 7);
  k_final<<<cdiv((long)NND * 32, 256), 256, 0, stream>>>(
      Fb, batch, sumsq, counts, xin, out_w, out_b, (float*)d_out);
}
